// ReadoutModel_31645319037307
// MI455X (gfx1250) — matrix operand order repaired and checked by reading, not compiled, not run on hardware
//
#include <hip/hip_runtime.h>
#include <stdint.h>

#define WPB 4              // waves per block
#define BLK (WPB * 32)

typedef __bf16 bf16;
typedef __attribute__((ext_vector_type(16))) __bf16 v16bf;
typedef __attribute__((ext_vector_type(8)))  __bf16 v8bf;
typedef __attribute__((ext_vector_type(8)))  float  v8f;

// ---------- small helpers ----------
__device__ __forceinline__ bf16 f2bf(float f) {
  union { float f; uint32_t u; } v; v.f = f;
  uint32_t r = v.u + 0x7FFFu + ((v.u >> 16) & 1u);   // round-to-nearest-even
  union { uint16_t u; bf16 b; } o; o.u = (uint16_t)(r >> 16);
  return o.b;
}
__device__ __forceinline__ float bf2f(bf16 b) {
  union { uint16_t u; bf16 b; } i; i.b = b;
  union { uint32_t u; float f; } o; o.u = ((uint32_t)i.u) << 16;
  return o.f;
}
__device__ __forceinline__ v8f vzero8() {
  v8f z;
#pragma unroll
  for (int i = 0; i < 8; ++i) z[i] = 0.f;
  return z;
}
__device__ __forceinline__ v16bf pack16(v8bf lo, v8bf hi) {
  v16bf a;
#pragma unroll
  for (int i = 0; i < 8; ++i) { a[i] = lo[i]; a[i + 8] = hi[i]; }
  return a;
}
// A fragment (16x32 bf16): lane holds row m = lane&15.
// Within a 32-wide K chunk: elems 0..15 -> K = hf*16+0..15 (contiguous)
__device__ __forceinline__ v16bf load_a(const bf16* rowChunk, int hf) {
  const v8bf* g0 = (const v8bf*)(rowChunk + hf * 16);
  const v8bf* g1 = (const v8bf*)(rowChunk + hf * 16 + 8);
  return pack16(*g0, *g1);
}
// B fragment (32x16 bf16, column-major per lane): lane holds col n, K = hf*16 + e (contiguous)
__device__ __forceinline__ v16bf load_b(const bf16* colPtr) {
  const v8bf* g0 = (const v8bf*)(colPtr);
  const v8bf* g1 = (const v8bf*)(colPtr + 8);
  return pack16(*g0, *g1);
}
__device__ __forceinline__ v8f wmma_bf16(v16bf a, v16bf b, v8f c) {
  return __builtin_amdgcn_wmma_f32_16x16x32_bf16(false, a, false, b, (short)0, c,
                                                 false, false);
}
// float atomic min/max via signed/unsigned int trick
__device__ __forceinline__ void atomicMaxF(float* a, float v) {
  if (__float_as_int(v) >= 0) atomicMax((int*)a, __float_as_int(v));
  else                        atomicMin((unsigned int*)a, __float_as_uint(v));
}
__device__ __forceinline__ void atomicMinF(float* a, float v) {
  if (__float_as_int(v) >= 0) atomicMin((int*)a, __float_as_int(v));
  else                        atomicMax((unsigned int*)a, __float_as_uint(v));
}

// ---------- tiny utility kernels ----------
__global__ void k_fill(float* p, float v, int n) {
  int i = blockIdx.x * blockDim.x + threadIdx.x;
  if (i < n) p[i] = v;
}
__global__ void k_count(const int* dst, float* cnt, int E) {
  int i = blockIdx.x * blockDim.x + threadIdx.x;
  if (i < E) atomicAdd(&cnt[dst[i]], 1.0f);
}
// emb = node_features(N,3) @ init_W(3,64) + init_b
__global__ void k_init_emb(const float* nf, const float* W, const float* b,
                           float* emb, int N) {
  int i = blockIdx.x * blockDim.x + threadIdx.x;
  if (i >= N * 64) return;
  int n = i >> 6, c = i & 63;
  float acc = b[c];
#pragma unroll
  for (int f = 0; f < 3; ++f) acc += nf[n * 3 + f] * W[f * 64 + c];
  emb[i] = acc;
}
// per-iteration prep: h = relu(emb), reset aggregation buffers
__global__ void k_prep_iter(const float* emb, float* h, float* sum, float* sumsq,
                            float* mn, float* mx, int total) {
  int i = blockIdx.x * blockDim.x + threadIdx.x;
  if (i >= total) return;
  h[i] = fmaxf(emb[i], 0.f);
  sum[i] = 0.f;
  sumsq[i] = 0.f;
  mn[i] = __int_as_float(0x7F800000);   // +inf
  mx[i] = __int_as_float(0xFF800000);   // -inf
}
// out[n*Kpad + k] = bf16(W[k*64 + n]) (transpose + zero-pad K)
__global__ void k_wT_bf16(const float* W, bf16* out, int K, int Kpad) {
  int i = blockIdx.x * blockDim.x + threadIdx.x;
  if (i >= 64 * Kpad) return;
  int n = i / Kpad, k = i % Kpad;
  out[i] = f2bf(k < K ? W[k * 64 + n] : 0.f);
}

// ---------- message kernel: per-edge MLP(130->64->64) fused with aggregation ----------
__global__ void __launch_bounds__(BLK)
k_msg(const float* __restrict__ hbuf, const float* __restrict__ ef,
      const int* __restrict__ srcIdx, const int* __restrict__ dstIdx,
      const bf16* __restrict__ W1t, const float* __restrict__ b1,
      const bf16* __restrict__ W2t, const float* __restrict__ b2,
      float* sum, float* sumsq, float* mn, float* mx, int E) {
  __shared__ __attribute__((aligned(16))) bf16 sX[WPB][16][160];
  __shared__ __attribute__((aligned(16))) bf16 sY[WPB][16][64];
  const int tid = threadIdx.x;
  const int w = tid >> 5, lane = tid & 31;
  const int m = lane & 15, hf = lane >> 4;
  const int numTiles = (E + 15) >> 4;

  for (int t0 = blockIdx.x * WPB; t0 < numTiles; t0 += gridDim.x * WPB) {
    const int tile = t0 + w;
    const int base = tile * 16;
    int r = base + m;
    int rc = (r < E) ? r : 0;
    int dc = dstIdx[rc], sc = srcIdx[rc];
    const float* hd = hbuf + (size_t)dc * 64;
    const float* hs = hbuf + (size_t)sc * 64;
#pragma unroll 4
    for (int ff = 0; ff < 80; ++ff) {
      int f = hf * 80 + ff;
      float v;
      if (f < 64)       v = hd[f];
      else if (f < 128) v = hs[f - 64];
      else if (f < 130) v = ef[(size_t)rc * 2 + (f - 128)];
      else              v = 0.f;
      sX[w][m][f] = f2bf(v);
    }
    __syncthreads();

    // layer 1: X(16x160) @ W1(160x64)
    v8f acc[4];
#pragma unroll
    for (int nt = 0; nt < 4; ++nt) acc[nt] = vzero8();
#pragma unroll
    for (int kc = 0; kc < 5; ++kc) {
      v16bf a = load_a(&sX[w][m][kc * 32], hf);
#pragma unroll
      for (int nt = 0; nt < 4; ++nt) {
        int n = nt * 16 + m;
        v16bf bb = load_b(W1t + (size_t)n * 160 + kc * 32 + hf * 16);
        acc[nt] = wmma_bf16(a, bb, acc[nt]);
      }
    }
    // bias + relu -> sY
#pragma unroll
    for (int nt = 0; nt < 4; ++nt) {
      int n = nt * 16 + m;
      float bv = b1[n];
#pragma unroll
      for (int e = 0; e < 8; ++e)
        sY[w][e + hf * 8][n] = f2bf(fmaxf(acc[nt][e] + bv, 0.f));
    }
    __syncthreads();

    // layer 2: Y(16x64) @ W2(64x64)
    v8f acc2[4];
#pragma unroll
    for (int nt = 0; nt < 4; ++nt) acc2[nt] = vzero8();
#pragma unroll
    for (int kc = 0; kc < 2; ++kc) {
      v16bf a = load_a(&sY[w][m][kc * 32], hf);
#pragma unroll
      for (int nt = 0; nt < 4; ++nt) {
        int n = nt * 16 + m;
        v16bf bb = load_b(W2t + (size_t)n * 64 + kc * 32 + hf * 16);
        acc2[nt] = wmma_bf16(a, bb, acc2[nt]);
      }
    }
    // aggregate into sum/sumsq/min/max via atomics (L2-resident node buffers)
    int drow[8];
#pragma unroll
    for (int e = 0; e < 8; ++e) {
      int rr = base + e + hf * 8;
      drow[e] = (rr < E) ? dstIdx[rr] : -1;
    }
#pragma unroll
    for (int nt = 0; nt < 4; ++nt) {
      int n = nt * 16 + m;
      float bv = b2[n];
#pragma unroll
      for (int e = 0; e < 8; ++e) {
        if (drow[e] >= 0) {
          float mv = acc2[nt][e] + bv;
          size_t o = (size_t)drow[e] * 64 + n;
          atomicAdd(&sum[o], mv);
          atomicAdd(&sumsq[o], mv * mv);
          atomicMinF(&mn[o], mv);
          atomicMaxF(&mx[o], mv);
        }
      }
    }
    __syncthreads();
  }
}

// ---------- update kernel: per-node MLP(320->64->64) ----------
__global__ void __launch_bounds__(BLK)
k_update(const float* __restrict__ hbuf, const float* __restrict__ cnt,
         const float* __restrict__ sum, const float* __restrict__ sumsq,
         const float* __restrict__ mnb, const float* __restrict__ mxb,
         const bf16* __restrict__ W1t, const float* __restrict__ b1,
         const bf16* __restrict__ W2t, const float* __restrict__ b2,
         float* emb, int N) {
  __shared__ __attribute__((aligned(16))) bf16 sX[WPB][16][320];
  __shared__ __attribute__((aligned(16))) bf16 sY[WPB][16][64];
  const int tid = threadIdx.x;
  const int w = tid >> 5, lane = tid & 31;
  const int m = lane & 15, hf = lane >> 4;
  const int numTiles = (N + 15) >> 4;

  for (int t0 = blockIdx.x * WPB; t0 < numTiles; t0 += gridDim.x * WPB) {
    const int tile = t0 + w;
    const int base = tile * 16;
    int r = base + m;
    int rc = (r < N) ? r : 0;
    float ct = cnt[rc];
    float ctc = fmaxf(ct, 1.f);
    bool he = ct > 0.f;
    const float* sR  = sum   + (size_t)rc * 64;
    const float* s2R = sumsq + (size_t)rc * 64;
    const float* mnR = mnb   + (size_t)rc * 64;
    const float* mxR = mxb   + (size_t)rc * 64;
    const float* hR  = hbuf  + (size_t)rc * 64;
#pragma unroll 4
    for (int ff = 0; ff < 160; ++ff) {
      int f = hf * 160 + ff;
      int c = f & 63, seg = f >> 6;
      float v;
      if (seg == 0) {
        float mu = sR[c] / ctc, m2 = s2R[c] / ctc;
        v = sqrtf(fmaxf(m2 - mu * mu, 0.f) + 1e-5f);
      } else if (seg == 1) v = he ? mnR[c] : 0.f;
      else if (seg == 2)   v = he ? mxR[c] : 0.f;
      else if (seg == 3)   v = sR[c] / ctc;
      else                 v = hR[c];
      sX[w][m][f] = f2bf(v);
    }
    __syncthreads();

    // layer 1: X(16x320) @ W1(320x64)
    v8f acc[4];
#pragma unroll
    for (int nt = 0; nt < 4; ++nt) acc[nt] = vzero8();
#pragma unroll
    for (int kc = 0; kc < 10; ++kc) {
      v16bf a = load_a(&sX[w][m][kc * 32], hf);
#pragma unroll
      for (int nt = 0; nt < 4; ++nt) {
        int n = nt * 16 + m;
        v16bf bb = load_b(W1t + (size_t)n * 320 + kc * 32 + hf * 16);
        acc[nt] = wmma_bf16(a, bb, acc[nt]);
      }
    }
#pragma unroll
    for (int nt = 0; nt < 4; ++nt) {
      int n = nt * 16 + m;
      float bv = b1[n];
#pragma unroll
      for (int e = 0; e < 8; ++e)
        sY[w][e + hf * 8][n] = f2bf(fmaxf(acc[nt][e] + bv, 0.f));
    }
    __syncthreads();

    // layer 2: Y(16x64) @ W2(64x64) -> new emb (no relu)
    v8f acc2[4];
#pragma unroll
    for (int nt = 0; nt < 4; ++nt) acc2[nt] = vzero8();
#pragma unroll
    for (int kc = 0; kc < 2; ++kc) {
      v16bf a = load_a(&sY[w][m][kc * 32], hf);
#pragma unroll
      for (int nt = 0; nt < 4; ++nt) {
        int n = nt * 16 + m;
        v16bf bb = load_b(W2t + (size_t)n * 64 + kc * 32 + hf * 16);
        acc2[nt] = wmma_bf16(a, bb, acc2[nt]);
      }
    }
#pragma unroll
    for (int nt = 0; nt < 4; ++nt) {
      int n = nt * 16 + m;
      float bv = b2[n];
#pragma unroll
      for (int e = 0; e < 8; ++e) {
        int rr = base + e + hf * 8;
        if (rr < N) emb[(size_t)rr * 64 + n] = acc2[nt][e] + bv;
      }
    }
    __syncthreads();
  }
}

// ---------- readout kernel: per-edge MLP(130->64->1) ----------
__global__ void __launch_bounds__(BLK)
k_readout(const float* __restrict__ emb, const float* __restrict__ ef,
          const int* __restrict__ srcIdx, const int* __restrict__ dstIdx,
          const bf16* __restrict__ W1t, const float* __restrict__ b1,
          const float* __restrict__ w2, const float* __restrict__ b2,
          float* out, int E) {
  __shared__ __attribute__((aligned(16))) bf16 sX[WPB][16][160];
  __shared__ __attribute__((aligned(16))) bf16 sY[WPB][16][64];
  const int tid = threadIdx.x;
  const int w = tid >> 5, lane = tid & 31;
  const int m = lane & 15, hf = lane >> 4;
  const int numTiles = (E + 15) >> 4;

  for (int t0 = blockIdx.x * WPB; t0 < numTiles; t0 += gridDim.x * WPB) {
    const int tile = t0 + w;
    const int base = tile * 16;
    int r = base + m;
    int rc = (r < E) ? r : 0;
    int sc = srcIdx[rc], dc = dstIdx[rc];
    const float* es = emb + (size_t)sc * 64;   // attacker = src first
    const float* ed = emb + (size_t)dc * 64;
#pragma unroll 4
    for (int ff = 0; ff < 80; ++ff) {
      int f = hf * 80 + ff;
      float v;
      if (f < 64)       v = es[f];
      else if (f < 128) v = ed[f - 64];
      else if (f < 130) v = ef[(size_t)rc * 2 + (f - 128)];
      else              v = 0.f;
      sX[w][m][f] = f2bf(v);
    }
    __syncthreads();

    v8f acc[4];
#pragma unroll
    for (int nt = 0; nt < 4; ++nt) acc[nt] = vzero8();
#pragma unroll
    for (int kc = 0; kc < 5; ++kc) {
      v16bf a = load_a(&sX[w][m][kc * 32], hf);
#pragma unroll
      for (int nt = 0; nt < 4; ++nt) {
        int n = nt * 16 + m;
        v16bf bb = load_b(W1t + (size_t)n * 160 + kc * 32 + hf * 16);
        acc[nt] = wmma_bf16(a, bb, acc[nt]);
      }
    }
#pragma unroll
    for (int nt = 0; nt < 4; ++nt) {
      int n = nt * 16 + m;
      float bv = b1[n];
#pragma unroll
      for (int e = 0; e < 8; ++e)
        sY[w][e + hf * 8][n] = f2bf(fmaxf(acc[nt][e] + bv, 0.f));
    }
    __syncthreads();

    // final 64->1 layer: lanes 0..15 each dot one row
    if (hf == 0) {
      int rr = base + m;
      if (rr < E) {
        float a2 = b2[0];
#pragma unroll 8
        for (int k = 0; k < 64; ++k) a2 += bf2f(sY[w][m][k]) * w2[k];
        out[rr] = a2;
      }
    }
    __syncthreads();
  }
}

// ---------- host side ----------
extern "C" void kernel_launch(void* const* d_in, const int* in_sizes, int n_in,
                              void* d_out, int out_size, void* d_ws, size_t ws_size,
                              hipStream_t stream) {
  const float* node_features = (const float*)d_in[0];
  const float* edge_features = (const float*)d_in[1];
  const float* init_W = (const float*)d_in[2];
  const float* init_b = (const float*)d_in[3];
  const float* msg_W1 = (const float*)d_in[4];
  const float* msg_b1 = (const float*)d_in[5];
  const float* msg_W2 = (const float*)d_in[6];
  const float* msg_b2 = (const float*)d_in[7];
  const float* upd_W1 = (const float*)d_in[8];
  const float* upd_b1 = (const float*)d_in[9];
  const float* upd_W2 = (const float*)d_in[10];
  const float* upd_b2 = (const float*)d_in[11];
  const float* ro_W1 = (const float*)d_in[12];
  const float* ro_b1 = (const float*)d_in[13];
  const float* ro_W2 = (const float*)d_in[14];
  const float* ro_b2 = (const float*)d_in[15];
  const int*   ei    = (const int*)d_in[16];

  const int N = in_sizes[0] / 3;
  const int E = in_sizes[16] / 2;
  const int* srcIdx = ei;
  const int* dstIdx = ei + E;

  // workspace layout (floats, then bf16 transposed weights)
  float* ws   = (float*)d_ws;
  float* cnt  = ws;
  float* emb  = cnt + N;
  float* h    = emb + (size_t)N * 64;
  float* sum  = h + (size_t)N * 64;
  float* sumsq= sum + (size_t)N * 64;
  float* mnb  = sumsq + (size_t)N * 64;
  float* mxb  = mnb + (size_t)N * 64;
  uintptr_t p = (uintptr_t)(mxb + (size_t)N * 64);
  p = (p + 15) & ~(uintptr_t)15;
  bf16* mW1t = (bf16*)p;              // 64 x 160
  bf16* mW2t = mW1t + 64 * 160;       // 64 x 64
  bf16* uW1t = mW2t + 64 * 64;        // 64 x 320
  bf16* uW2t = uW1t + 64 * 320;       // 64 x 64
  bf16* rW1t = uW2t + 64 * 64;        // 64 x 160

  const int NC = N * 64;
  k_fill<<<(N + 255) / 256, 256, 0, stream>>>(cnt, 0.f, N);
  k_count<<<(E + 255) / 256, 256, 0, stream>>>(dstIdx, cnt, E);
  k_init_emb<<<(NC + 255) / 256, 256, 0, stream>>>(node_features, init_W, init_b, emb, N);
  k_wT_bf16<<<(64 * 160 + 255) / 256, 256, 0, stream>>>(msg_W1, mW1t, 130, 160);
  k_wT_bf16<<<(64 * 64 + 255) / 256, 256, 0, stream>>>(msg_W2, mW2t, 64, 64);
  k_wT_bf16<<<(64 * 320 + 255) / 256, 256, 0, stream>>>(upd_W1, uW1t, 320, 320);
  k_wT_bf16<<<(64 * 64 + 255) / 256, 256, 0, stream>>>(upd_W2, uW2t, 64, 64);
  k_wT_bf16<<<(64 * 160 + 255) / 256, 256, 0, stream>>>(ro_W1, rW1t, 130, 160);

  const int msgBlocks = 1920;
  const int updBlocks = 800;
  for (int it = 0; it < 5; ++it) {
    k_prep_iter<<<(NC + 255) / 256, 256, 0, stream>>>(emb, h, sum, sumsq, mnb, mxb, NC);
    k_msg<<<msgBlocks, BLK, 0, stream>>>(h, edge_features, srcIdx, dstIdx,
                                         mW1t, msg_b1, mW2t, msg_b2,
                                         sum, sumsq, mnb, mxb, E);
    k_update<<<updBlocks, BLK, 0, stream>>>(h, cnt, sum, sumsq, mnb, mxb,
                                            uW1t, upd_b1, uW2t, upd_b2, emb, N);
  }
  k_readout<<<msgBlocks, BLK, 0, stream>>>(emb, edge_features, srcIdx, dstIdx,
                                           rW1t, ro_b1, ro_W2, ro_b2,
                                           (float*)d_out, E);
}
